// Net3DLayer_75058848465486
// MI455X (gfx1250) — matrix operand order repaired and checked by reading, not compiled, not run on hardware
//
#include <hip/hip_runtime.h>
#include <math.h>

#define N_NODES 50000
#define N_EDGES 800000
#define HID 128

typedef __attribute__((ext_vector_type(16))) _Float16 v16h;
typedef __attribute__((ext_vector_type(8)))  float    v8f;

union V16HQ { v16h v; uint4 q[2]; };

__device__ __forceinline__ v8f wmma_f16(v16h a, v16h b, v8f c) {
  // D = A(16x32 f16) * B(32x16 f16) + C(16x16 f32)
  return __builtin_amdgcn_wmma_f32_16x16x32_f16(
      /*neg_a=*/false, a, /*neg_b=*/false, b,
      /*c_mod=*/(short)0, c, /*reuse_a=*/false, /*reuse_b=*/false);
}

// B fragment: group0 at p, group1 at p+16 (lane n = l%16, K = 8*(l>=16) + h, +16 for h>=8)
__device__ __forceinline__ v16h load_b_lds(const _Float16* p) {
  V16HQ u;
  u.q[0] = *(const uint4*)(p);
  u.q[1] = *(const uint4*)(p + 16);
  return u.v;
}

// A fragment from f16 LDS row: group0 at p, group1 at p+16
__device__ __forceinline__ v16h load_a_lds(const _Float16* p) {
  V16HQ u;
  u.q[0] = *(const uint4*)(p);
  u.q[1] = *(const uint4*)(p + 16);
  return u.v;
}

__device__ __forceinline__ v16h cvt8x2(float4 f0, float4 f1, float4 f2, float4 f3) {
  v16h a;
  a[0]=(_Float16)f0.x;  a[1]=(_Float16)f0.y;  a[2]=(_Float16)f0.z;  a[3]=(_Float16)f0.w;
  a[4]=(_Float16)f1.x;  a[5]=(_Float16)f1.y;  a[6]=(_Float16)f1.z;  a[7]=(_Float16)f1.w;
  a[8]=(_Float16)f2.x;  a[9]=(_Float16)f2.y;  a[10]=(_Float16)f2.z; a[11]=(_Float16)f2.w;
  a[12]=(_Float16)f3.x; a[13]=(_Float16)f3.y; a[14]=(_Float16)f3.z; a[15]=(_Float16)f3.w;
  return a;
}

__device__ __forceinline__ float4 ld4(const float* p) { return *(const float4*)p; }
__device__ __forceinline__ float4 add4(float4 a, float4 b) {
  return make_float4(a.x + b.x, a.y + b.y, a.z + b.z, a.w + b.w);
}
__device__ __forceinline__ float silu_f(float x) { return x / (1.f + __expf(-x)); }

// ---------------- weight transpose + f32->f16 convert ----------------
// dst[n*K + k] = (f16) src[k*N + n]
__global__ void cvt_t(const float* __restrict__ s, _Float16* __restrict__ d, int K, int N) {
  int i = blockIdx.x * blockDim.x + threadIdx.x;
  if (i >= K * N) return;
  int n = i / K, k = i - n * K;
  d[i] = (_Float16)s[(size_t)k * N + n];
}

// ---------------- edge kernel ----------------
#define EDGE_WAVES 4
#define EDGE_TPW 4

__global__ __launch_bounds__(128) void edge_kernel(
    const float* __restrict__ feat, const float* __restrict__ dmat,
    const int* __restrict__ src, const int* __restrict__ dst,
    const _Float16* __restrict__ wts,   // wt1m(49152) | wt2m(16384) halves, contiguous
    const float* __restrict__ b1m, const float* __restrict__ b2m,
    const float* __restrict__ Wg, const float* __restrict__ bg,
    float* __restrict__ msum, float* __restrict__ dout) {
  extern __shared__ __align__(32) char smem[];
  _Float16* wt1m = (_Float16*)smem;            // [128][384] f16, 98304 B
  _Float16* wt2m = wt1m + 49152;               // [128][128] f16, 32768 B
  float* b1s = (float*)(smem + 131072);        // 128 f32
  float* b2s = b1s + 128;
  float* wgs = b2s + 128;
  _Float16* hbuf = (_Float16*)(smem + 132608); // 4 waves * [16][128] f16

  // stage weights (131072 B) + biases into LDS
  {
    const uint4* gs = (const uint4*)wts;
    uint4* ds = (uint4*)wt1m;
    for (int i = threadIdx.x; i < 8192; i += 128) ds[i] = gs[i];
    if (threadIdx.x < 128) {
      b1s[threadIdx.x] = b1m[threadIdx.x];
      b2s[threadIdx.x] = b2m[threadIdx.x];
      wgs[threadIdx.x] = Wg[threadIdx.x];
    }
    __syncthreads();
  }

  const int lane = threadIdx.x & 31;
  const int wave = threadIdx.x >> 5;
  const int m    = lane & 15;      // A row / B col / D col within tile
  const int hgrp = lane >> 4;      // lane half (0/1)
  const int kg   = hgrp * 8;       // A-fragment K sub-offset
  const float bgv = bg[0];
  _Float16* hb = hbuf + wave * 2048;

  for (int t = 0; t < EDGE_TPW; ++t) {
    const int tile  = (blockIdx.x * EDGE_WAVES + wave) * EDGE_TPW + t;
    const int ebase = tile * 16;
    const int erow  = ebase + m;
    const int si = src[erow];
    const int di = dst[erow];

    v8f acc[8];
#pragma unroll
    for (int nt = 0; nt < 8; ++nt)
#pragma unroll
      for (int v = 0; v < 8; ++v) acc[nt][v] = 0.f;

    // ---- layer 1: [16x384] @ [384x128], K-tiles align with concat segments
    for (int kt = 0; kt < 12; ++kt) {
      const float* p;
      if (kt < 4)      p = feat + (size_t)si   * HID + kt * 32;
      else if (kt < 8) p = feat + (size_t)di   * HID + (kt - 4) * 32;
      else             p = dmat + (size_t)erow * HID + (kt - 8) * 32;
      v16h a = cvt8x2(ld4(p + kg), ld4(p + kg + 4), ld4(p + 16 + kg), ld4(p + 16 + kg + 4));
#pragma unroll
      for (int nt = 0; nt < 8; ++nt) {
        v16h b = load_b_lds(wt1m + (nt * 16 + m) * 384 + kt * 32 + kg);
        acc[nt] = wmma_f16(a, b, acc[nt]);
      }
    }

    // ---- bias + SiLU, spill h as f16 to wave-private LDS (M-major)
#pragma unroll
    for (int nt = 0; nt < 8; ++nt) {
      const float bias = b1s[nt * 16 + m];
#pragma unroll
      for (int v = 0; v < 8; ++v) {
        float x = acc[nt][v] + bias;
        hb[(v + hgrp * 8) * HID + nt * 16 + m] = (_Float16)silu_f(x);
      }
    }

    // ---- layer 2: [16x128] @ [128x128]
    v8f mg[8];
#pragma unroll
    for (int nt = 0; nt < 8; ++nt)
#pragma unroll
      for (int v = 0; v < 8; ++v) mg[nt][v] = 0.f;
    for (int kt = 0; kt < 4; ++kt) {
      v16h a = load_a_lds(hb + m * HID + kt * 32 + kg);
#pragma unroll
      for (int nt = 0; nt < 8; ++nt) {
        v16h b = load_b_lds(wt2m + (nt * 16 + m) * 128 + kt * 32 + kg);
        mg[nt] = wmma_f16(a, b, mg[nt]);
      }
    }

    // ---- bias + SiLU -> message; gate partials (dot with Wg over this lane's 8 cols)
    float part[8];
#pragma unroll
    for (int v = 0; v < 8; ++v) part[v] = 0.f;
#pragma unroll
    for (int nt = 0; nt < 8; ++nt) {
      const float bias = b2s[nt * 16 + m];
      const float wgv  = wgs[nt * 16 + m];
#pragma unroll
      for (int v = 0; v < 8; ++v) {
        float s = silu_f(mg[nt][v] + bias);
        mg[nt][v] = s;
        part[v] += s * wgv;
      }
    }
    // reduce across the 16 lanes of this half (xor masks stay within the half)
    float gate[8];
#pragma unroll
    for (int v = 0; v < 8; ++v) {
      float p = part[v];
      p += __shfl_xor(p, 1);
      p += __shfl_xor(p, 2);
      p += __shfl_xor(p, 4);
      p += __shfl_xor(p, 8);
      gate[v] = 1.f / (1.f + __expf(-(p + bgv)));
    }

    int drow[8];
#pragma unroll
    for (int v = 0; v < 8; ++v) drow[v] = dst[ebase + v + hgrp * 8];

    // ---- residual edge output + gated atomic scatter into node accumulator
#pragma unroll
    for (int nt = 0; nt < 8; ++nt) {
      const int col = nt * 16 + m;
#pragma unroll
      for (int v = 0; v < 8; ++v) {
        const int e = ebase + v + hgrp * 8;
        const float ms = mg[nt][v];
        dout[(size_t)e * HID + col] = dmat[(size_t)e * HID + col] + ms;
        atomicAdd(msum + (size_t)drow[v] * HID + col, ms * gate[v]);
      }
    }
  }
}

// ---------------- node kernel ----------------
#define NODE_WAVES 4
#define NODE_TPW 5

__global__ __launch_bounds__(128) void node_kernel(
    const float* __restrict__ feat,
    const _Float16* __restrict__ wts,  // wt1u(16384) | wt2u(16384) halves
    const float* __restrict__ b1u, const float* __restrict__ b2u,
    float* __restrict__ featio) {      // in: m_sum ; out: feat_out
  extern __shared__ __align__(32) char smem[];
  _Float16* wt1u = (_Float16*)smem;
  _Float16* wt2u = wt1u + 16384;
  float* b1s = (float*)(smem + 65536);
  float* b2s = b1s + 128;
  _Float16* hbuf = (_Float16*)(smem + 66560);

  {
    const uint4* gs = (const uint4*)wts;
    uint4* ds = (uint4*)wt1u;
    for (int i = threadIdx.x; i < 4096; i += 128) ds[i] = gs[i];
    if (threadIdx.x < 128) {
      b1s[threadIdx.x] = b1u[threadIdx.x];
      b2s[threadIdx.x] = b2u[threadIdx.x];
    }
    __syncthreads();
  }

  const int lane = threadIdx.x & 31;
  const int wave = threadIdx.x >> 5;
  const int m    = lane & 15;
  const int hgrp = lane >> 4;
  const int kg   = hgrp * 8;
  _Float16* hb = hbuf + wave * 2048;

  for (int t = 0; t < NODE_TPW; ++t) {
    const int tile = (blockIdx.x * NODE_WAVES + wave) * NODE_TPW + t;
    if (tile >= N_NODES / 16) break;
    const int nbase = tile * 16;
    const int nrow  = nbase + m;

    v8f acc[8];
#pragma unroll
    for (int nt = 0; nt < 8; ++nt)
#pragma unroll
      for (int v = 0; v < 8; ++v) acc[nt][v] = 0.f;

    // ---- layer 1: A = (m_sum + feat) fused on load
    for (int kt = 0; kt < 4; ++kt) {
      const float* pf = feat   + (size_t)nrow * HID + kt * 32;
      const float* pm = featio + (size_t)nrow * HID + kt * 32;
      v16h a = cvt8x2(add4(ld4(pf + kg),       ld4(pm + kg)),
                      add4(ld4(pf + kg + 4),   ld4(pm + kg + 4)),
                      add4(ld4(pf + 16 + kg),  ld4(pm + 16 + kg)),
                      add4(ld4(pf + 16 + kg + 4), ld4(pm + 16 + kg + 4)));
#pragma unroll
      for (int nt = 0; nt < 8; ++nt) {
        v16h b = load_b_lds(wt1u + (nt * 16 + m) * 128 + kt * 32 + kg);
        acc[nt] = wmma_f16(a, b, acc[nt]);
      }
    }

#pragma unroll
    for (int nt = 0; nt < 8; ++nt) {
      const float bias = b1s[nt * 16 + m];
#pragma unroll
      for (int v = 0; v < 8; ++v) {
        float x = acc[nt][v] + bias;
        hb[(v + hgrp * 8) * HID + nt * 16 + m] = (_Float16)silu_f(x);
      }
    }

    // ---- layer 2
    v8f o[8];
#pragma unroll
    for (int nt = 0; nt < 8; ++nt)
#pragma unroll
      for (int v = 0; v < 8; ++v) o[nt][v] = 0.f;
    for (int kt = 0; kt < 4; ++kt) {
      v16h a = load_a_lds(hb + m * HID + kt * 32 + kg);
#pragma unroll
      for (int nt = 0; nt < 8; ++nt) {
        v16h b = load_b_lds(wt2u + (nt * 16 + m) * 128 + kt * 32 + kg);
        o[nt] = wmma_f16(a, b, o[nt]);
      }
    }

    // ---- bias + residual, overwrite accumulator region with feat_out
#pragma unroll
    for (int nt = 0; nt < 8; ++nt) {
      const int col = nt * 16 + m;
      const float bias = b2s[col];
#pragma unroll
      for (int v = 0; v < 8; ++v) {
        const int r = nbase + v + hgrp * 8;
        featio[(size_t)r * HID + col] = o[nt][v] + bias + feat[(size_t)r * HID + col];
      }
    }
  }
}

// ---------------- host ----------------
extern "C" void kernel_launch(void* const* d_in, const int* in_sizes, int n_in,
                              void* d_out, int out_size, void* d_ws, size_t ws_size,
                              hipStream_t stream) {
  (void)in_sizes; (void)n_in; (void)out_size; (void)ws_size;
  const float* feat = (const float*)d_in[0];
  const float* dmat = (const float*)d_in[1];
  const int*   src  = (const int*)d_in[2];
  const int*   dst  = (const int*)d_in[3];
  const float* W1m  = (const float*)d_in[4];
  const float* b1m  = (const float*)d_in[5];
  const float* W2m  = (const float*)d_in[6];
  const float* b2m  = (const float*)d_in[7];
  const float* Wg   = (const float*)d_in[8];
  const float* bg   = (const float*)d_in[9];
  const float* W1u  = (const float*)d_in[10];
  const float* b1u  = (const float*)d_in[11];
  const float* W2u  = (const float*)d_in[12];
  const float* b2u  = (const float*)d_in[13];

  float* feat_out = (float*)d_out;                         // doubles as m_sum accumulator
  float* dout_e   = feat_out + (size_t)N_NODES * HID;

  _Float16* ws16 = (_Float16*)d_ws;
  _Float16* wt1m = ws16;              // 49152 halves [128][384]
  _Float16* wt2m = ws16 + 49152;      // 16384 halves [128][128]
  _Float16* wt1u = ws16 + 65536;      // 16384
  _Float16* wt2u = ws16 + 81920;      // 16384

  hipMemsetAsync(d_out, 0, (size_t)N_NODES * HID * sizeof(float), stream);
  cvt_t<<<(384 * 128 + 255) / 256, 256, 0, stream>>>(W1m, wt1m, 384, 128);
  cvt_t<<<(128 * 128 + 255) / 256, 256, 0, stream>>>(W2m, wt2m, 128, 128);
  cvt_t<<<(128 * 128 + 255) / 256, 256, 0, stream>>>(W1u, wt1u, 128, 128);
  cvt_t<<<(128 * 128 + 255) / 256, 256, 0, stream>>>(W2u, wt2u, 128, 128);

  const size_t lds_edge = 131072 + 3 * 128 * 4 + 4 * 2048 * 2;  // 148992 B
  const int edge_blocks = N_EDGES / (16 * EDGE_WAVES * EDGE_TPW); // 3125
  edge_kernel<<<edge_blocks, 128, lds_edge, stream>>>(
      feat, dmat, src, dst, wt1m, b1m, b2m, Wg, bg, feat_out, dout_e);

  const size_t lds_node = 65536 + 2 * 128 * 4 + 4 * 2048 * 2;    // 82944 B
  const int node_tiles = N_NODES / 16;                            // 3125
  const int node_blocks = (node_tiles + NODE_WAVES * NODE_TPW - 1) / (NODE_WAVES * NODE_TPW);
  node_kernel<<<node_blocks, 128, lds_node, stream>>>(feat, wt1u, b1u, b2u, feat_out);
}
